// MultiHeadAttention_14577119003176
// MI455X (gfx1250) — matrix operand order repaired and checked by reading, not compiled, not run on hardware
//
#include <hip/hip_runtime.h>

#define BB 4
#define SS 2048
#define DDM 1024
#define HH 16
#define DKV 64

typedef __attribute__((ext_vector_type(16))) _Float16 v16h;
typedef __attribute__((ext_vector_type(8)))  _Float16 v8h;
typedef __attribute__((ext_vector_type(8)))  float    v8f;
typedef __attribute__((ext_vector_type(4)))  unsigned v4u;
typedef __attribute__((ext_vector_type(8)))  int      v8i;
typedef __attribute__((ext_vector_type(4)))  int      v4i;

union AFU { v16h v; v8h h[2]; };

// ---- CDNA5 async global->LDS copy (ASYNCcnt path) ----------------------
// On AMDGPU a generic pointer into LDS carries the LDS byte offset in its
// low 32 bits, which is exactly what the VDST operand of the async-to-LDS
// instructions expects (an LDS offset added to the LDS base and the immediate offset).
static __device__ __forceinline__ void async_b128(const void* gsrc, void* ldst) {
  const unsigned loff = (unsigned)(uintptr_t)ldst;
  asm volatile("global_load_async_to_lds_b128 %0, %1, off"
               :: "v"(loff), "v"(gsrc) : "memory");
}
static __device__ __forceinline__ void async_wait_all() {
  asm volatile("s_wait_asynccnt 0x0" ::: "memory");
}

// A-fragment (16x32 f16, MxK): lane = (half, m); VGPR0-3 hold K=half*8..+7,
// VGPR4-7 hold K=16+half*8..+7  -> two contiguous 16B reads per lane.
static __device__ __forceinline__ v16h lds_afrag(const _Float16* base, int strideH, int lane) {
  const int g = lane >> 4;
  const int m = lane & 15;
  const _Float16* p = base + m * strideH + g * 8;
  AFU u;
  u.h[0] = *(const v8h*)(p);
  u.h[1] = *(const v8h*)(p + 16);
  return u.v;
}

// B-fragment (32x16 f16, KxN) from LDS stored transposed as [N][K]:
// lane = (half, n); VGPR0-3 hold K=half*8..+7, VGPR4-7 hold K=16+half*8..+7.
static __device__ __forceinline__ v16h lds_bfrag(const _Float16* base, int strideH, int lane) {
  const int g = lane >> 4;
  const int n = lane & 15;
  const _Float16* p = base + n * strideH + g * 8;
  AFU u;
  u.h[0] = *(const v8h*)(p);
  u.h[1] = *(const v8h*)(p + 16);
  return u.v;
}

#define WMMA16(a, b, c) \
  __builtin_amdgcn_wmma_f32_16x16x32_f16(false, (a), false, (b), (short)0, (c), false, false)

// ---------------------------------------------------------------------------
// Tiled GEMM: Out(MxN) = A(MxK) * W(KxN) + bias, optional row-mask multiply.
// Block = 256 threads (8 waves), 128x128 output tile, K stepped by 32.
// ---------------------------------------------------------------------------
template <bool A_F32, bool OUT_F32>
__global__ __launch_bounds__(256) void gemm128(const void* __restrict__ Ap,
                                               const float* __restrict__ W,
                                               const float* __restrict__ bias,
                                               void* __restrict__ Outp,
                                               const int* __restrict__ rmask,
                                               int M, int N, int Kd) {
  __shared__ __align__(16) _Float16 sA[128][40];  // [m][k], padded rows
  __shared__ __align__(16) _Float16 sB[128][40];  // [n][k] (W transposed)

  const int tid  = threadIdx.x;
  const int lane = tid & 31;
  const int wv   = tid >> 5;
  const int m0   = blockIdx.y * 128;
  const int n0   = blockIdx.x * 128;
  const int wm   = (wv >> 1) * 32;  // wave row offset (4 row-waves)
  const int wn   = (wv & 1) * 64;   // wave col offset (2 col-waves)

  v8f zero = {};
  v8f acc[2][4];
#pragma unroll
  for (int i = 0; i < 2; ++i)
#pragma unroll
    for (int j = 0; j < 4; ++j) acc[i][j] = zero;

  const int row  = tid >> 1;        // 0..127
  const int koff = (tid & 1) * 16;  // 0 or 16

  for (int k0 = 0; k0 < Kd; k0 += 32) {
    __syncthreads();
    // ---- stage A tile (128x32) as f16 ----
    if (A_F32) {
      const float* A = (const float*)Ap;
      const float4* src = (const float4*)(A + (size_t)(m0 + row) * Kd + k0 + koff);
#pragma unroll
      for (int c = 0; c < 4; ++c) {
        float4 f = src[c];
        sA[row][koff + c * 4 + 0] = (_Float16)f.x;
        sA[row][koff + c * 4 + 1] = (_Float16)f.y;
        sA[row][koff + c * 4 + 2] = (_Float16)f.z;
        sA[row][koff + c * 4 + 3] = (_Float16)f.w;
      }
      if (k0 + 32 < Kd)
        __builtin_prefetch(A + (size_t)(m0 + row) * Kd + k0 + 32 + koff, 0, 1);
    } else {
      // f16 A: pure copy -> async global->LDS path (no VGPR round trip)
      const _Float16* A = (const _Float16*)Ap;
      const _Float16* src = A + (size_t)(m0 + row) * Kd + k0 + koff;
      async_b128(src, &sA[row][koff]);
      async_b128(src + 8, &sA[row][koff + 8]);
      if (k0 + 32 < Kd)
        __builtin_prefetch(src + 32, 0, 1);
    }
    // ---- stage W tile transposed: sB[n][k] = W[k0+k][n0+n] ----
    {
      const int n  = row;
      const int ks = koff;
#pragma unroll
      for (int j = 0; j < 16; ++j)
        sB[n][ks + j] = (_Float16)W[(size_t)(k0 + ks + j) * N + n0 + n];
      if (k0 + 32 < Kd)
        __builtin_prefetch(W + (size_t)(k0 + 32 + ks) * N + n0 + n, 0, 1);
    }
    if (!A_F32) async_wait_all();
    __syncthreads();

    v16h aF0 = lds_afrag(&sA[wm][0], 40, lane);
    v16h aF1 = lds_afrag(&sA[wm + 16][0], 40, lane);
#pragma unroll
    for (int j = 0; j < 4; ++j) {
      v16h bF = lds_bfrag(&sB[wn + j * 16][0], 40, lane);
      acc[0][j] = WMMA16(aF0, bF, acc[0][j]);
      acc[1][j] = WMMA16(aF1, bF, acc[1][j]);
    }
  }

  // ---- epilogue: bias (+ mask) and store ----
  const int g = lane >> 4;
  const int n = lane & 15;
#pragma unroll
  for (int j = 0; j < 4; ++j) {
    const int col = n0 + wn + j * 16 + n;
    const float bc = bias[col];
#pragma unroll
    for (int i = 0; i < 2; ++i) {
#pragma unroll
      for (int e = 0; e < 8; ++e) {
        const int rowg = m0 + wm + i * 16 + e + 8 * g;
        float val = acc[i][j][e] + bc;
        if (OUT_F32) {
          val *= (float)rmask[rowg];
          ((float*)Outp)[(size_t)rowg * N + col] = val;
        } else {
          ((_Float16*)Outp)[(size_t)rowg * N + col] = (_Float16)val;
        }
      }
    }
  }
}

// ---------------------------------------------------------------------------
// Flash attention: one block per (b, h, 128 q-rows); each wave owns 16 rows.
// Online softmax with causal + v_mask masking. All matmuls via WMMA f16.
// Q tile staged by the Tensor Data Mover; K tiles via async global->LDS.
// ---------------------------------------------------------------------------
__global__ __launch_bounds__(256) void attn_fwd(const _Float16* __restrict__ Qp,
                                                const _Float16* __restrict__ Kp,
                                                const _Float16* __restrict__ Vp,
                                                const int* __restrict__ vmask,
                                                _Float16* __restrict__ Op) {
  __shared__ __align__(16) _Float16 sQ[128][72];   // [q][dk], 128B rows + 16B pad
  __shared__ __align__(16) _Float16 sK[32][72];    // [key][dk]  (== B^T layout)
  __shared__ __align__(16) _Float16 sV[64][40];    // [dv][key]  (V transposed)
  __shared__ __align__(16) float    sS[8][16][33]; // per-wave raw scores
  __shared__ __align__(16) _Float16 sP[8][16][40]; // per-wave exp(P) f16
  __shared__ float sStat[8][16];                   // per-wave alpha / 1/l

  const int tid  = threadIdx.x;
  const int lane = tid & 31;
  const int wv   = tid >> 5;
  const int g    = lane >> 4;
  const int ln   = lane & 15;
  const int q0   = blockIdx.x * 128;
  const int h    = blockIdx.y;
  const int b    = blockIdx.z;
  const size_t hoff = (size_t)h * DKV;

  // ---- stage Q tile (128 rows x 128B) once, via TDM when available ----
#if __has_builtin(__builtin_amdgcn_tensor_load_to_lds)
  if (wv == 0) {
    const uintptr_t ga = (uintptr_t)(Qp + (size_t)(b * SS + q0) * (HH * DKV) + hoff);
    const unsigned  la = (unsigned)(uintptr_t)&sQ[0][0];
    // D# group0: count=1 user descriptor, lds_addr, global_addr[56:0], type=2
    v4u g0 = { 1u, la, (unsigned)ga,
               ((unsigned)(ga >> 32) & 0x01FFFFFFu) | (2u << 30) };
    // D# group1: data_size=2B, pad_enable, pad_interval=32dw(128B),
    // pad_amount=4dw(16B); tensor_dim0=64, tensor_dim1=128;
    // tile 64x128; tensor_dim0_stride=1024 elems (H*DKV row)
    v8i g1 = { (int)((1u << 16) | (1u << 20) | (4u << 22) | (3u << 25)),
               (int)(64u << 16),          // barrier_addr=0 | tensor_dim0.lo
               (int)(128u << 16),         // tensor_dim0.hi=0 | tensor_dim1.lo
               (int)(64u << 16),          // tensor_dim1.hi=0 | tile_dim0=64
               (int)128,                  // tile_dim1=128 | tile_dim2=0
               (int)(HH * DKV),           // tensor_dim0_stride.lo = 1024
               0, 0 };
    v4i g2 = { 1, 0, 0, 0 };             // tensor_dim2=1 (unused)
    v4i g3 = { 0, 0, 0, 0 };
    v8i g4 = { 0, 0, 0, 0, 0, 0, 0, 0 }; // extra group (6-arg toolchain form)
    __builtin_amdgcn_tensor_load_to_lds(g0, g1, g2, g3, g4, 0);
    __builtin_amdgcn_s_wait_tensorcnt(0);
  }
#else
  {
    const int r   = tid >> 1;
    const int off = (tid & 1) * 32;
    const v8h* src = (const v8h*)(Qp + (size_t)(b * SS + q0 + r) * (HH * DKV) + hoff + off);
    *(v8h*)&sQ[r][off]      = src[0];
    *(v8h*)&sQ[r][off + 8]  = src[1];
    *(v8h*)&sQ[r][off + 16] = src[2];
    *(v8h*)&sQ[r][off + 24] = src[3];
  }
#endif
  __syncthreads();

  const int qbase = q0 + wv * 16;
  const v16h aQ0 = lds_afrag(&sQ[wv * 16][0], 72, lane);
  const v16h aQ1 = lds_afrag(&sQ[wv * 16][32], 72, lane);

  v8f zero = {};
  v8f accO[4];
#pragma unroll
  for (int t = 0; t < 4; ++t) accO[t] = zero;
  float mrun = -1e30f, lrun = 0.f;

  const int nkt = (q0 + 128) / 32;  // causal: keys needed up to q0+127
  for (int kt = 0; kt < nkt; ++kt) {
    const int k0 = kt * 32;
    __syncthreads();
    // stage K tile (32 keys x 64 dk) via async global->LDS, 16B per lane
    {
      const int key = tid >> 3, doff = (tid & 7) * 8;
      const _Float16* src =
          Kp + (size_t)(b * SS + k0 + key) * (HH * DKV) + hoff + doff;
      async_b128(src, &sK[key][doff]);
    }
    // stage V tile transposed: sV[dv][key] (transpose needs the VGPR path)
    {
      const int key = tid & 31, dv = (tid >> 5) * 8;
      const v8h src = *(const v8h*)(Vp + (size_t)(b * SS + k0 + key) * (HH * DKV) + hoff + dv);
#pragma unroll
      for (int jj = 0; jj < 8; ++jj) sV[dv + jj][key] = src[jj];
    }
    async_wait_all();
    __syncthreads();

    if (k0 <= qbase + 15) {  // wave has at least one unmasked (q,k) pair
      // scores S = Q * K^T (16x32), K-dim 64 -> 2 chunks
      v8f s0 = zero, s1 = zero;
      {
        v16h b00 = lds_bfrag(&sK[0][0],  72, lane);
        v16h b01 = lds_bfrag(&sK[16][0], 72, lane);
        s0 = WMMA16(aQ0, b00, s0);
        s1 = WMMA16(aQ0, b01, s1);
        v16h b10 = lds_bfrag(&sK[0][32],  72, lane);
        v16h b11 = lds_bfrag(&sK[16][32], 72, lane);
        s0 = WMMA16(aQ1, b10, s0);
        s1 = WMMA16(aQ1, b11, s1);
      }
      const int vm0 = vmask[b * SS + k0 + ln];
      const int vm1 = vmask[b * SS + k0 + 16 + ln];
      const float scl = 0.125f;  // 1/sqrt(64)
#pragma unroll
      for (int e = 0; e < 8; ++e) {
        const int gq = qbase + e + 8 * g;  // C layout: row = e + 8*half
        float x0 = s0[e] * scl;
        float x1 = s1[e] * scl;
        if ((k0 + ln) > gq || vm0 == 0) x0 = -1e12f;
        if ((k0 + 16 + ln) > gq || vm1 == 0) x1 = -1e12f;
        sS[wv][e + 8 * g][ln]      = x0;
        sS[wv][e + 8 * g][16 + ln] = x1;
      }
      // online-softmax row stats (lanes 0..15, one row each; same-wave LDS
      // ops are in-order so no barrier needed for this wave-private strip)
      if (lane < 16) {
        const float* rw = &sS[wv][lane][0];
        float tmax = -1e30f;
#pragma unroll
        for (int j2 = 0; j2 < 32; ++j2) tmax = fmaxf(tmax, rw[j2]);
        const float mnew = fmaxf(mrun, tmax);
        const float al = __expf(mrun - mnew);
        float sum = 0.f;
#pragma unroll
        for (int j2 = 0; j2 < 32; ++j2) {
          const float p = __expf(rw[j2] - mnew);
          sum += p;
          sP[wv][lane][j2] = (_Float16)p;
        }
        lrun = lrun * al + sum;
        mrun = mnew;
        sStat[wv][lane] = al;
      }
      // rescale accumulators and add P * V
      float al8[8];
#pragma unroll
      for (int e = 0; e < 8; ++e) al8[e] = sStat[wv][e + 8 * g];
      const v16h pF = lds_afrag(&sP[wv][0][0], 40, lane);
#pragma unroll
      for (int t = 0; t < 4; ++t) {
#pragma unroll
        for (int e = 0; e < 8; ++e) accO[t][e] *= al8[e];
        v16h bV = lds_bfrag(&sV[t * 16][0], 40, lane);
        accO[t] = WMMA16(pF, bV, accO[t]);
      }
    }
  }

  // normalize by softmax denom and store O (f16)
  if (lane < 16) sStat[wv][lane] = 1.f / lrun;
  float inv8[8];
#pragma unroll
  for (int e = 0; e < 8; ++e) inv8[e] = sStat[wv][e + 8 * g];
#pragma unroll
  for (int t = 0; t < 4; ++t)
#pragma unroll
    for (int e = 0; e < 8; ++e) {
      const int gq = qbase + e + 8 * g;
      Op[(size_t)(b * SS + gq) * (HH * DKV) + hoff + t * 16 + ln] =
          (_Float16)(accO[t][e] * inv8[e]);
    }
}

// ---------------------------------------------------------------------------
extern "C" void kernel_launch(void* const* d_in, const int* in_sizes, int n_in,
                              void* d_out, int out_size, void* d_ws, size_t ws_size,
                              hipStream_t stream) {
  (void)in_sizes; (void)n_in; (void)out_size; (void)ws_size;
  const float* q      = (const float*)d_in[0];
  const float* k      = (const float*)d_in[1];
  const float* v      = (const float*)d_in[2];
  const int*   q_mask = (const int*)d_in[3];
  const int*   v_mask = (const int*)d_in[4];
  const float* Wq     = (const float*)d_in[5];
  const float* bq     = (const float*)d_in[6];
  const float* Wk     = (const float*)d_in[7];
  const float* bk     = (const float*)d_in[8];
  const float* Wv     = (const float*)d_in[9];
  const float* bv     = (const float*)d_in[10];
  const float* Wo     = (const float*)d_in[11];
  const float* bo     = (const float*)d_in[12];

  const int M = BB * SS;      // 8192
  const int N = HH * DKV;     // 1024
  const int K = DDM;          // 1024

  const size_t planeElems = (size_t)M * N;  // f16 elems per plane (16 MB)
  _Float16* Qp    = (_Float16*)d_ws;
  _Float16* Kp    = Qp + planeElems;
  _Float16* Vp    = Kp + planeElems;
  _Float16* Oattn = Vp + planeElems;

  dim3 gGemm(N / 128, M / 128);  // (8, 64)
  dim3 bDim(256);

  // Q/K/V projections: f32 activations -> f16 outputs, WMMA accumulation
  gemm128<true, false><<<gGemm, bDim, 0, stream>>>((const void*)q, Wq, bq, (void*)Qp, nullptr, M, N, K);
  gemm128<true, false><<<gGemm, bDim, 0, stream>>>((const void*)k, Wk, bk, (void*)Kp, nullptr, M, N, K);
  gemm128<true, false><<<gGemm, bDim, 0, stream>>>((const void*)v, Wv, bv, (void*)Vp, nullptr, M, N, K);

  // flash attention per (b, h, 128-row q tile)
  dim3 gAttn(SS / 128, HH, BB);  // (16, 16, 4)
  attn_fwd<<<gAttn, bDim, 0, stream>>>(Qp, Kp, Vp, v_mask, Oattn);

  // output projection + q_mask, f32 output
  gemm128<false, true><<<gGemm, bDim, 0, stream>>>((const void*)Oattn, Wo, bo, d_out, q_mask, M, N, K);
}
